// PGBF_Surv_75411035783468
// MI455X (gfx1250) — matrix operand order repaired and checked by reading, not compiled, not run on hardware
//
#include <hip/hip_runtime.h>
#include <math.h>

// ---------------------------------------------------------------- types
typedef __attribute__((ext_vector_type(16))) __bf16        v16bf;
typedef __attribute__((ext_vector_type(8)))  float         v8f;
typedef __attribute__((ext_vector_type(8)))  unsigned int  v8u;

union Frag16 { uint4 q[2]; v16bf b; };         // 32 bytes = 16 bf16
union FragB  { v8u u; v16bf b; };

static __device__ __forceinline__ unsigned short f2bf(float f) {
    unsigned int u = __float_as_uint(f);
    unsigned int r = u + 0x7FFFu + ((u >> 16) & 1u);   // RNE
    return (unsigned short)(r >> 16);
}

#define FLAG_RELU  1
#define FLAG_F32   2
#define FLAG_BF16  4
#define FLAG_ACCUM 8

// ---------------------------------------------------------------- fp32 -> bf16
__global__ void cvt_bf16_kernel(const float* __restrict__ in,
                                unsigned short* __restrict__ out, int n) {
    int i = blockIdx.x * blockDim.x + threadIdx.x;
    if (i < n) out[i] = f2bf(in[i]);
}

// ------------------------------------------------ pack weight [K,N] into WMMA-B fragments
// layout: frag[(kt*ntiles + nt)*32 + lane][16] ; lane n=lane&15, hi=lane>>4 holds
// B[kbase + hi*16 + t][nbase + n] for t=0..15 (K-contiguous per lane)
__global__ void pack_b_kernel(const float* __restrict__ W,
                              unsigned short* __restrict__ out, int K, int N) {
    int lane = threadIdx.x;                 // 32 threads
    int nt = blockIdx.x, kt = blockIdx.y;
    int n  = (nt << 4) + (lane & 15);
    int kb = (kt << 5) + ((lane >> 4) << 4);
    unsigned short* dst = out + (((size_t)kt * gridDim.x + nt) * 32 + lane) * 16;
    #pragma unroll
    for (int t = 0; t < 16; ++t)
        dst[t] = f2bf(W[(size_t)(kb + t) * N + n]);
}

// ---------------------------------------------------------------- generic WMMA GEMM
// C[M,N] = epilogue(A_bf16[M,K] @ Bpacked[K,N] + bias)
// block = 256 thr = 8 waves; wave -> 16x16 tile; block tile = 128 rows x 16 cols
__global__ __launch_bounds__(256) void gemm_bf16_kernel(
        const unsigned short* __restrict__ A, const unsigned short* __restrict__ Bp,
        const float* __restrict__ bias, float* __restrict__ Cf,
        unsigned short* __restrict__ Cb, int M, int K, int N, int flags, float bscale) {
    int lane  = threadIdx.x & 31;
    int wave  = threadIdx.x >> 5;
    int rbase = blockIdx.y * 128 + wave * 16;
    int ct    = blockIdx.x;                     // column tile
    int m     = lane & 15, hi = lane >> 4;
    int ntiles = N >> 4;
    const unsigned short* arow = A + (size_t)(rbase + m) * K;

    v8f acc = {};
    int ksteps = K >> 5;
    for (int kt = 0; kt < ksteps; ++kt) {
        int kb = kt << 5;
        Frag16 fa;
        fa.q[0] = *(const uint4*)(arow + kb + hi * 16);
        fa.q[1] = *(const uint4*)(arow + kb + hi * 16 + 8);
        FragB fb;
        fb.u = *(const v8u*)(Bp + (((size_t)kt * ntiles + ct) * 32 + lane) * 16);
        acc = __builtin_amdgcn_wmma_f32_16x16x32_bf16(
                  false, fa.b, false, fb.b, (short)0, acc, false, false);
    }

    int col = (ct << 4) + m;
    float bv = bias[col];
    #pragma unroll
    for (int r = 0; r < 8; ++r) {
        int row = rbase + r + hi * 8;
        float v = acc[r] + bv;
        if (flags & FLAG_RELU) v = v > 0.f ? v : 0.f;
        size_t off = (size_t)row * N + col;
        if (flags & FLAG_ACCUM) v += Cf[off];
        if (flags & FLAG_F32)  Cf[off] = v;
        if (flags & FLAG_BF16) Cb[off] = f2bf(v * bscale);
    }
}

// ------------------------------------- fused attn logits (WMMA) + top-12 + softmax
// EhS: e_h * SCALE in bf16 [N,256]; Etb: e_t bf16 [N,256]
// Wave handles 16 rows. The block stages each 16-row e_t tile (8KB) into LDS ONCE
// via CDNA5 async global->LDS DMA (ASYNCcnt), double-buffered, shared by 8 waves.
__global__ __launch_bounds__(256) void attn_topk_kernel(
        const unsigned short* __restrict__ EhS, const unsigned short* __restrict__ Etb,
        float* __restrict__ tkp, int* __restrict__ tki, int Nrows) {
    __shared__ __attribute__((aligned(32))) unsigned short bstage[2][16 * 256]; // 2 x 8KB
    __shared__ float tile[8][16][17];           // per-wave 16x16 logit tile (padded)
    int lane  = threadIdx.x & 31;
    int wave  = threadIdx.x >> 5;
    int rbase = (blockIdx.x * 8 + wave) * 16;
    int m = lane & 15, hi = lane >> 4;

    // preload A fragments for these 16 rows (K = 256 -> 8 ksteps), resident in VGPRs
    v16bf afrag[8];
    {
        const unsigned short* arow = EhS + (size_t)(rbase + m) * 256;
        #pragma unroll
        for (int kt = 0; kt < 8; ++kt) {
            Frag16 fa;
            fa.q[0] = *(const uint4*)(arow + kt * 32 + hi * 16);
            fa.q[1] = *(const uint4*)(arow + kt * 32 + hi * 16 + 8);
            afrag[kt] = fa.b;
        }
    }

    float tv[12]; int ti[12];
    #pragma unroll
    for (int j = 0; j < 12; ++j) { tv[j] = -3.4e38f; ti[j] = 0; }

    // async staging helpers: each thread moves 16B at linear and linear+4096
    // (INST_OFFSET applies to both LDS and global addresses per ISA 10.7 pseudocode)
    unsigned linear = (unsigned)threadIdx.x * 16u;                    // byte offset in tile
    unsigned ldsb0  = (unsigned)(size_t)&bstage[0][0] + linear;
    unsigned ldsb1  = (unsigned)(size_t)&bstage[1][0] + linear;

#define ISSUE_STAGE(T, LDSADDR)                                                   \
    do {                                                                          \
        const char* gbase_ = (const char*)(Etb + ((size_t)(T) << 4) * 256);       \
        asm volatile("global_load_async_to_lds_b128 %0, %1, %2 offset:0"          \
                     :: "v"(LDSADDR), "v"(linear), "s"(gbase_) : "memory");       \
        asm volatile("global_load_async_to_lds_b128 %0, %1, %2 offset:4096"       \
                     :: "v"(LDSADDR), "v"(linear), "s"(gbase_) : "memory");       \
    } while (0)

    int ctiles = Nrows >> 4;
    ISSUE_STAGE(0, ldsb0);
    for (int ct = 0; ct < ctiles; ++ct) {
        int cur = ct & 1;
        if (ct + 1 < ctiles) {
            ISSUE_STAGE(ct + 1, (cur ? ldsb0 : ldsb1));
            asm volatile("s_wait_asynccnt 0x2" ::: "memory");   // tile ct done, ct+1 in flight
        } else {
            asm volatile("s_wait_asynccnt 0x0" ::: "memory");
        }
        __syncthreads();                        // tile ct visible to all 8 waves

        int cb = ct << 4;
        v8f acc = {};
        const unsigned short* bb = &bstage[cur][m * 256 + hi * 16];
        #pragma unroll
        for (int kt = 0; kt < 8; ++kt) {
            FragB fb;
            fb.u = *(const v8u*)(bb + kt * 32);          // ds_load_b128 x2
            acc = __builtin_amdgcn_wmma_f32_16x16x32_bf16(
                      false, afrag[kt], false, fb.b, (short)0, acc, false, false);
        }
        #pragma unroll
        for (int r = 0; r < 8; ++r)
            tile[wave][r + hi * 8][m] = acc[r];
        __builtin_amdgcn_wave_barrier();
        if (lane < 16) {                        // lane owns row rbase+lane
            #pragma unroll
            for (int c = 0; c < 16; ++c) {
                float v = tile[wave][lane][c];
                if (v > tv[11]) {               // insert into sorted-descending list
                    float cv = v; int ci = cb + c;
                    #pragma unroll
                    for (int j = 0; j < 12; ++j) {
                        bool g = cv > tv[j];
                        float nv = g ? tv[j] : cv;  int ni = g ? ti[j] : ci;
                        tv[j]    = g ? cv : tv[j];  ti[j]  = g ? ci : ti[j];
                        cv = nv; ci = ni;
                    }
                }
            }
        }
        __syncthreads();                        // all waves done with bstage[cur]
    }
#undef ISSUE_STAGE

    if (lane < 16) {                            // softmax over top-12 + emit
        int row = rbase + lane;
        float mx = tv[0], s = 0.f, e[12];
        #pragma unroll
        for (int j = 0; j < 12; ++j) { e[j] = __expf(tv[j] - mx); s += e[j]; }
        float inv = 1.f / s;
        #pragma unroll
        for (int j = 0; j < 12; ++j) {
            tkp[(size_t)row * 12 + j] = e[j] * inv;
            tki[(size_t)row * 12 + j] = ti[j];
        }
    }
}

// --------------------------- per-row aggregation: gate, ka-softmax, e_Nh, s/b emit
__global__ __launch_bounds__(256) void aggregate_kernel(
        const float* __restrict__ ehf, const float* __restrict__ etf,
        const float* __restrict__ tkp, const int* __restrict__ tki,
        unsigned short* __restrict__ sbf, unsigned short* __restrict__ bbf) {
    __shared__ float red[8][12];
    int row = blockIdx.x, d = threadIdx.x;
    float eh = ehf[(size_t)row * 256 + d];
    float p[12], nb[12], w[12]; int ix[12];
    #pragma unroll
    for (int k = 0; k < 12; ++k) {
        p[k]  = tkp[(size_t)row * 12 + k];
        ix[k] = tki[(size_t)row * 12 + k];
    }
    #pragma unroll
    for (int k = 0; k < 12; ++k) {
        nb[k] = etf[(size_t)ix[k] * 256 + d];
        float ehr = p[k] * nb[k] + (1.f - p[k]) * eh;
        w[k] = nb[k] * tanhf(eh + ehr);
    }
    // wave32 reduce, then cross-wave via LDS
    #pragma unroll
    for (int k = 0; k < 12; ++k)
        for (int msk = 16; msk >= 1; msk >>= 1)
            w[k] += __shfl_xor(w[k], msk, 32);
    int wv = threadIdx.x >> 5, ln = threadIdx.x & 31;
    if (ln == 0) {
        #pragma unroll
        for (int k = 0; k < 12; ++k) red[wv][k] = w[k];
    }
    __syncthreads();
    float kw[12];
    #pragma unroll
    for (int k = 0; k < 12; ++k) {
        float s = 0.f;
        #pragma unroll
        for (int q = 0; q < 8; ++q) s += red[q][k];
        kw[k] = s;
    }
    float mx = kw[0];
    #pragma unroll
    for (int k = 1; k < 12; ++k) mx = fmaxf(mx, kw[k]);
    float ssum = 0.f, kp[12];
    #pragma unroll
    for (int k = 0; k < 12; ++k) { kp[k] = __expf(kw[k] - mx); ssum += kp[k]; }
    float inv = 1.f / ssum, enh = 0.f;
    #pragma unroll
    for (int k = 0; k < 12; ++k) enh += kp[k] * inv * nb[k];
    sbf[(size_t)row * 256 + d] = f2bf(eh + enh);
    bbf[(size_t)row * 256 + d] = f2bf(eh * enh);
}

// ---------------------------------------------------------------- launcher
extern "C" void kernel_launch(void* const* d_in, const int* in_sizes, int n_in,
                              void* d_out, int out_size, void* d_ws, size_t ws_size,
                              hipStream_t stream) {
    const float* x     = (const float*)d_in[0];
    const float* wsi_w = (const float*)d_in[1];
    const float* wsi_b = (const float*)d_in[2];
    const float* wh_w  = (const float*)d_in[3];
    const float* wh_b  = (const float*)d_in[4];
    const float* wt_w  = (const float*)d_in[5];
    const float* wt_b  = (const float*)d_in[6];
    const float* w1_w  = (const float*)d_in[7];
    const float* w1_b  = (const float*)d_in[8];
    const float* w2_w  = (const float*)d_in[9];
    const float* w2_b  = (const float*)d_in[10];
    float* out = (float*)d_out;

    const int N = 12288, DIN = 1024, D = 256;
    const float SCALE = 0.0625f;   // 256^-0.5

    char* ws = (char*)d_ws;
    size_t off = 0;
    auto alloc = [&](size_t bytes) -> char* {
        char* p = ws + off;
        off += (bytes + 255) & ~(size_t)255;
        return p;
    };
    unsigned short* x_bf = (unsigned short*)alloc((size_t)N * DIN * 2);
    unsigned short* pwsi = (unsigned short*)alloc((size_t)DIN * D * 2);
    unsigned short* pwh  = (unsigned short*)alloc((size_t)D * D * 2);
    unsigned short* pwt  = (unsigned short*)alloc((size_t)D * D * 2);
    unsigned short* pw1  = (unsigned short*)alloc((size_t)D * D * 2);
    unsigned short* pw2  = (unsigned short*)alloc((size_t)D * D * 2);
    unsigned short* h_bf = (unsigned short*)alloc((size_t)N * D * 2);
    float*          ehf  = (float*)alloc((size_t)N * D * 4);
    float*          etf  = (float*)alloc((size_t)N * D * 4);
    unsigned short* ehb  = (unsigned short*)alloc((size_t)N * D * 2);  // pre-scaled
    unsigned short* etb  = (unsigned short*)alloc((size_t)N * D * 2);
    float*          tkp  = (float*)alloc((size_t)N * 12 * 4);
    int*            tki  = (int*)alloc((size_t)N * 12 * 4);
    unsigned short* s_bf = (unsigned short*)alloc((size_t)N * D * 2);
    unsigned short* b_bf = (unsigned short*)alloc((size_t)N * D * 2);

    // 1) x -> bf16
    cvt_bf16_kernel<<<(N * DIN) / 256, 256, 0, stream>>>(x, x_bf, N * DIN);
    // 2) pack weights into WMMA-B fragment order
    pack_b_kernel<<<dim3(D / 16, DIN / 32), 32, 0, stream>>>(wsi_w, pwsi, DIN, D);
    pack_b_kernel<<<dim3(D / 16, D / 32),   32, 0, stream>>>(wh_w,  pwh,  D, D);
    pack_b_kernel<<<dim3(D / 16, D / 32),   32, 0, stream>>>(wt_w,  pwt,  D, D);
    pack_b_kernel<<<dim3(D / 16, D / 32),   32, 0, stream>>>(w1_w,  pw1,  D, D);
    pack_b_kernel<<<dim3(D / 16, D / 32),   32, 0, stream>>>(w2_w,  pw2,  D, D);
    // 3) h = relu(x @ wsi + b)  (bf16)
    gemm_bf16_kernel<<<dim3(D / 16, N / 128), 256, 0, stream>>>(
        x_bf, pwsi, wsi_b, nullptr, h_bf, N, DIN, D, FLAG_RELU | FLAG_BF16, 1.f);
    // 4) e_h (fp32 + bf16*SCALE), e_t (fp32 + bf16)
    gemm_bf16_kernel<<<dim3(D / 16, N / 128), 256, 0, stream>>>(
        h_bf, pwh, wh_b, ehf, ehb, N, D, D, FLAG_F32 | FLAG_BF16, SCALE);
    gemm_bf16_kernel<<<dim3(D / 16, N / 128), 256, 0, stream>>>(
        h_bf, pwt, wt_b, etf, etb, N, D, D, FLAG_F32 | FLAG_BF16, 1.f);
    // 5) fused attention logits + top-12 + softmax (async LDS staging)
    attn_topk_kernel<<<N / 128, 256, 0, stream>>>(ehb, etb, tkp, tki, N);
    // 6) gate / ka-softmax / e_Nh -> s, b (bf16)
    aggregate_kernel<<<N, 256, 0, stream>>>(ehf, etf, tkp, tki, s_bf, b_bf);
    // 7) out = relu(s@w1+b1) + relu(b@w2+b2)
    gemm_bf16_kernel<<<dim3(D / 16, N / 128), 256, 0, stream>>>(
        s_bf, pw1, w1_b, out, nullptr, N, D, D, FLAG_RELU | FLAG_F32, 1.f);
    gemm_bf16_kernel<<<dim3(D / 16, N / 128), 256, 0, stream>>>(
        b_bf, pw2, w2_b, out, nullptr, N, D, D, FLAG_RELU | FLAG_F32 | FLAG_ACCUM, 1.f);
}
